// DKVB_62354335203617
// MI455X (gfx1250) — matrix operand order repaired and checked by reading, not compiled, not run on hardware
//
#include <hip/hip_runtime.h>
#include <hip/hip_bf16.h>

// ---------------------------------------------------------------------------
// Problem constants (from reference): B=4, N=64 -> 256 tokens, D=2048, C=256,
// E=128, M=4096, V=10.
// ---------------------------------------------------------------------------
#define TOKENS 256
#define D_DIM  2048
#define C_DIM  256
#define E_DIM  128
#define M_DIM  4096
#define V_DIM  10
#define MT     128           // codebook rows staged per LDS chunk in dist kernel
#define CBS    (E_DIM + 8)   // padded LDS row stride (bf16 units) to dodge bank conflicts

typedef __attribute__((ext_vector_type(16))) __bf16 v16bf;
typedef __attribute__((ext_vector_type(4)))  __bf16 v4bf;
typedef __attribute__((ext_vector_type(8)))  float  v8f;

// ---------------------------------------------------------------------------
// WMMA fragment loaders (wave32); contiguous K order k = 16h + e (this code's choice, not the ISA table order).
//
// 16-bit A matrix 16x32 (MxK): lane L -> row M = L&15; K base = (L>>4)*16;
//   element e holds K = kbase + e (same order as B).
// 16-bit B matrix 32x16 (KxN): lane L -> col N = L&15; K base = (L>>4)*16;
//   element e holds K = kbase + e.
// f32 C/D 16x16: vgpr r, lane L -> row M = r + ((L>>4)*8), col N = L&15.
// ---------------------------------------------------------------------------
__device__ __forceinline__ v16bf load_A_f32(const float* __restrict__ base,
                                            int row0, int ld, int k0) {
  const int lane = threadIdx.x & 31;
  const int m    = lane & 15;
  const int koff = (lane >> 4) * 16;
  const float* rp = base + (size_t)(row0 + m) * ld + k0 + koff;
  v16bf a;
#pragma unroll
  for (int i = 0; i < 8; ++i) {
    const int k = 2 * i;
    a[2 * i]     = (__bf16)rp[k];
    a[2 * i + 1] = (__bf16)rp[k + 1];
  }
  return a;
}

__device__ __forceinline__ v16bf load_A_bf16(const __bf16* __restrict__ base,
                                             int row0, int ld, int k0) {
  const int lane = threadIdx.x & 31;
  const int m    = lane & 15;
  const int koff = (lane >> 4) * 16;
  const __bf16* rp = base + (size_t)(row0 + m) * ld + k0 + koff;
  v16bf a;
#pragma unroll
  for (int i = 0; i < 8; ++i) {
    const int k = 2 * i;
    a[2 * i]     = rp[k];
    a[2 * i + 1] = rp[k + 1];
  }
  return a;
}

// B(K,N) = src[(k0+K)*ldn + n0 + N]   (source row-major over K, used in K1)
__device__ __forceinline__ v16bf load_B_strided_f32(const float* __restrict__ base,
                                                    int k0, int ldn, int n0) {
  const int lane = threadIdx.x & 31;
  const int n  = lane & 15;
  const int kb = (lane >> 4) * 16;
  const float* p = base + (size_t)(k0 + kb) * ldn + n0 + n;
  v16bf b;
#pragma unroll
  for (int e = 0; e < 16; ++e) b[e] = (__bf16)p[(size_t)e * ldn];
  return b;
}

// B(K,N) = cb[(m0+N)*ld + k0 + K]   (transposed source: codebook rows = N, used in K2)
__device__ __forceinline__ v16bf load_B_lds(const __bf16* cb, int m0, int ld, int k0) {
  const int lane = threadIdx.x & 31;
  const int n  = lane & 15;
  const int kb = (lane >> 4) * 16;
  const __bf16* rp = cb + (size_t)(m0 + n) * ld + k0 + kb;
  v16bf b;
#pragma unroll
  for (int e = 0; e < 16; ++e) b[e] = rp[e];
  return b;
}

// ---------------------------------------------------------------------------
// Kernel 1: xp[c,t,e] = sum_d emb[t,d] * proj[c,d,e]   (stored bf16)
// grid = C, block = 256 (8 waves). Wave w owns E-column tile [16w,16w+16).
// Each wave keeps all 16 token-tile accumulators in registers so each
// rand_proj element is read exactly once per workgroup.
// ---------------------------------------------------------------------------
__global__ void __launch_bounds__(256)
proj_kernel(const float* __restrict__ emb, const float* __restrict__ proj,
            __bf16* __restrict__ xp) {
  const int c    = blockIdx.x;
  const int lane = threadIdx.x & 31;
  const int w    = threadIdx.x >> 5;   // 0..7
  const int n0   = w * 16;
  const float* projc = proj + (size_t)c * D_DIM * E_DIM;

  v8f acc[16];
#pragma unroll
  for (int t = 0; t < 16; ++t) acc[t] = v8f{};

  for (int ks = 0; ks < D_DIM / 32; ++ks) {
    const int k0 = ks * 32;
    v16bf b = load_B_strided_f32(projc, k0, E_DIM, n0);
#pragma unroll
    for (int t = 0; t < 16; ++t) {
      v16bf a = load_A_f32(emb, t * 16, D_DIM, k0);
      acc[t] = __builtin_amdgcn_wmma_f32_16x16x32_bf16(
          false, a, false, b, (short)0, acc[t], false, false);
    }
  }

  const int n     = lane & 15;
  const int rhalf = (lane >> 4) * 8;
#pragma unroll
  for (int t = 0; t < 16; ++t) {
#pragma unroll
    for (int r = 0; r < 8; ++r) {
      const int row = t * 16 + rhalf + r;
      xp[((size_t)c * TOKENS + row) * E_DIM + n0 + n] = (__bf16)acc[t][r];
    }
  }
}

// ---------------------------------------------------------------------------
// Kernel 2: per (c): scores = e_sq[m] - 2 * (xp_c @ codebook_c^T); argmin over m.
// grid = C, block = 256 (8 waves). Wave w owns tokens [32w, 32w+32).
// Codebook chunk (128 rows x 128 E, fp32) staged to LDS as bf16 by all 256
// threads; per-row e_sq computed in the same pass (no extra HBM traffic).
// ---------------------------------------------------------------------------
__global__ void __launch_bounds__(256)
dist_kernel(const float* __restrict__ codebook, const __bf16* __restrict__ xp,
            int* __restrict__ idx_out) {
  __shared__ __bf16 cb[MT * CBS];   // ~34 KB
  __shared__ float  esq[MT];
  __shared__ float  part[256];

  const int c    = blockIdx.x;
  const int tid  = threadIdx.x;
  const int lane = tid & 31;
  const int w    = tid >> 5;
  const float*  cbc = codebook + (size_t)c * M_DIM * E_DIM;
  const __bf16* xpc = xp + (size_t)c * TOKENS * E_DIM;

  // Preload A fragments: 2 token tiles x 4 K-steps (whole K=128) live in regs.
  v16bf a[2][4];
#pragma unroll
  for (int tt = 0; tt < 2; ++tt)
#pragma unroll
    for (int ks = 0; ks < 4; ++ks)
      a[tt][ks] = load_A_bf16(xpc, w * 32 + tt * 16, E_DIM, ks * 32);

  float minv[2][8];
  int   mini[2][8];
#pragma unroll
  for (int tt = 0; tt < 2; ++tt)
#pragma unroll
    for (int r = 0; r < 8; ++r) { minv[tt][r] = 3.0e38f; mini[tt][r] = 0; }

  const int lrow = tid >> 1;          // 0..127: codebook row within chunk
  const int le0  = (tid & 1) * 64;    // which 64-element half of E

  for (int mc = 0; mc < M_DIM / MT; ++mc) {
    __syncthreads();  // previous chunk's compute done before overwrite
    const float* src = cbc + ((size_t)(mc * MT + lrow)) * E_DIM + le0;
    float s = 0.f;
#pragma unroll
    for (int j = 0; j < 16; ++j) {
      float4 f = ((const float4*)src)[j];
      s += f.x * f.x + f.y * f.y + f.z * f.z + f.w * f.w;
      v4bf pk;
      pk[0] = (__bf16)f.x; pk[1] = (__bf16)f.y;
      pk[2] = (__bf16)f.z; pk[3] = (__bf16)f.w;
      *(v4bf*)&cb[(size_t)lrow * CBS + le0 + 4 * j] = pk;
    }
    part[tid] = s;
    __syncthreads();
    if (tid < MT) esq[tid] = part[2 * tid] + part[2 * tid + 1];
    __syncthreads();

    const int n = lane & 15;
    for (int ct = 0; ct < MT / 16; ++ct) {
      v16bf b[4];
#pragma unroll
      for (int ks = 0; ks < 4; ++ks) b[ks] = load_B_lds(cb, ct * 16, CBS, ks * 32);
      const float eq    = esq[ct * 16 + n];
      const int   mglob = mc * MT + ct * 16 + n;
#pragma unroll
      for (int tt = 0; tt < 2; ++tt) {
        v8f accv = v8f{};
#pragma unroll
        for (int ks = 0; ks < 4; ++ks)
          accv = __builtin_amdgcn_wmma_f32_16x16x32_bf16(
              false, a[tt][ks], false, b[ks], (short)0, accv, false, false);
#pragma unroll
        for (int r = 0; r < 8; ++r) {
          const float sc = eq - 2.0f * accv[r];
          if (sc < minv[tt][r]) { minv[tt][r] = sc; mini[tt][r] = mglob; }
        }
      }
    }
  }

  // Reduce argmin across the 16 lanes (N dimension) of each half-wave.
  // Tie-break toward lower index to match jnp.argmin semantics.
#pragma unroll
  for (int tt = 0; tt < 2; ++tt) {
#pragma unroll
    for (int r = 0; r < 8; ++r) {
      float v = minv[tt][r];
      int   i = mini[tt][r];
#pragma unroll
      for (int off = 1; off < 16; off <<= 1) {
        float ov = __shfl_xor(v, off, 32);
        int   oi = __shfl_xor(i, off, 32);
        if (ov < v || (ov == v && oi < i)) { v = ov; i = oi; }
      }
      if ((lane & 15) == 0) {
        const int row = w * 32 + tt * 16 + ((lane >> 4) * 8) + r;
        idx_out[(size_t)c * TOKENS + row] = i;
      }
    }
  }
}

// ---------------------------------------------------------------------------
// Kernel 3: out[t,v] = mean_c values[c, idx[c,t], v]
// grid = 256 tokens, block = 256 (one thread per c); deterministic LDS tree
// reduction (graph-replay safe, writes d_out fully).
// ---------------------------------------------------------------------------
__global__ void __launch_bounds__(256)
gather_mean_kernel(const float* __restrict__ values, const int* __restrict__ idx,
                   float* __restrict__ out) {
  __shared__ float partial[256][V_DIM];
  const int token = blockIdx.x;
  const int c     = threadIdx.x;

  const int m = idx[(size_t)c * TOKENS + token];
  const float* vp = values + ((size_t)c * M_DIM + m) * V_DIM;
#pragma unroll
  for (int v = 0; v < V_DIM; ++v) partial[c][v] = vp[v];
  __syncthreads();

  for (int s = 128; s > 0; s >>= 1) {
    if (c < s) {
#pragma unroll
      for (int v = 0; v < V_DIM; ++v) partial[c][v] += partial[c + s][v];
    }
    __syncthreads();
  }
  if (c == 0) {
#pragma unroll
    for (int v = 0; v < V_DIM; ++v)
      out[(size_t)token * V_DIM + v] = partial[0][v] * (1.0f / C_DIM);
  }
}

// ---------------------------------------------------------------------------
extern "C" void kernel_launch(void* const* d_in, const int* in_sizes, int n_in,
                              void* d_out, int out_size, void* d_ws, size_t ws_size,
                              hipStream_t stream) {
  const float* emb      = (const float*)d_in[0];  // (B,N,D)   = (256,2048)
  const float* proj     = (const float*)d_in[1];  // (C,D,E)
  const float* codebook = (const float*)d_in[2];  // (C,M,E)
  const float* values   = (const float*)d_in[3];  // (C,M,V)
  float* out = (float*)d_out;                     // (B,N,V)   = 2560 floats

  __bf16* xp  = (__bf16*)d_ws;                                    // 16 MB
  int*    idx = (int*)((char*)d_ws +
                       (size_t)C_DIM * TOKENS * E_DIM * sizeof(__bf16));  // 256 KB

  proj_kernel<<<C_DIM, 256, 0, stream>>>(emb, proj, xp);
  dist_kernel<<<C_DIM, 256, 0, stream>>>(codebook, xp, idx);
  gather_mean_kernel<<<TOKENS, 256, 0, stream>>>(values, idx, out);
}
